// LlamaAttention_PALU_60481729462616
// MI455X (gfx1250) — matrix operand order repaired and checked by reading, not compiled, not run on hardware
//
#include <hip/hip_runtime.h>
#include <hip/hip_bf16.h>

// ---------------------------------------------------------------------------
// Problem constants (fixed by the reference)
// ---------------------------------------------------------------------------
#define BB    2
#define SS    2048
#define HID   4096
#define NH    32
#define HD    128          // head dim
#define GG    8            // palu groups
#define RK    256          // rank per K group
#define FGD   256          // fused group dim (V rank per group)
#define FH    8192         // fused hidden dim

typedef __bf16 bf16;
typedef __attribute__((ext_vector_type(16))) __bf16 v16bf;
typedef __attribute__((ext_vector_type(8)))  __bf16 v8bf;
typedef __attribute__((ext_vector_type(8)))  float  v8f;

static __device__ __forceinline__ v16bf make_v16(v8bf lo, v8bf hi) {
  union { v16bf v; v8bf h[2]; } u;
  u.h[0] = lo; u.h[1] = hi;
  return u.v;
}

static __device__ __forceinline__ v8f zero8() {
  v8f z = {0.f, 0.f, 0.f, 0.f, 0.f, 0.f, 0.f, 0.f};
  return z;
}

// gfx1250 async global->LDS copy, 16B per lane, tracked by ASYNCcnt (§08).
// lds_off is the byte offset inside the workgroup LDS allocation (generic LDS
// pointers keep the offset in addr[31:0], so truncation recovers it).
static __device__ __forceinline__ void async_load_b128(unsigned int lds_off, const void* gptr) {
  asm volatile("global_load_async_to_lds_b128 %0, %1, off"
               :: "v"(lds_off), "v"((unsigned long long)(uintptr_t)gptr)
               : "memory");
}
static __device__ __forceinline__ void wait_async0() {
  asm volatile("s_wait_asynccnt 0" ::: "memory");
}
static __device__ __forceinline__ unsigned int lds_off_of(const void* p) {
  return (unsigned int)(uintptr_t)p;   // addr[31:0] of the LDS aperture address
}

// ---------------------------------------------------------------------------
// fp32 -> bf16 conversion (grid-stride)
// ---------------------------------------------------------------------------
__global__ void cvt_f32_bf16(const float* __restrict__ src, bf16* __restrict__ dst, long n) {
  long i = (long)blockIdx.x * blockDim.x + threadIdx.x;
  long stride = (long)gridDim.x * blockDim.x;
  for (; i < n; i += stride) dst[i] = (bf16)src[i];
}

// ---------------------------------------------------------------------------
// Generic bf16 GEMM:  C(M,N) = A(M,K) * B(N,K)^T, f32 accumulate.
// Tiles: BM=128, BN=128, BK=32.  256 threads = 8 waves in a 2(m) x 4(n) grid;
// each wave owns a 64x32 sub-tile = 4x2 WMMA fragments.
// Double-buffered LDS filled with GLOBAL_LOAD_ASYNC_TO_LDS_B128; one barrier
// per K-step (arrival proves the previous buffer is no longer being read).
// Optional batching over gridDim.z with element strides.
// ---------------------------------------------------------------------------
template <bool F32OUT>
__global__ __launch_bounds__(256) void gemm_bf16_nt(
    const bf16* __restrict__ A, int lda, int strideA,
    const bf16* __restrict__ B, int ldb, int strideB,
    void* __restrict__ Cptr, int ldc, int strideC,
    int K)
{
  __shared__ __align__(16) bf16 shA[2][128 * 32];
  __shared__ __align__(16) bf16 shB[2][128 * 32];

  const int bz = blockIdx.z;
  A += (size_t)bz * (size_t)strideA;
  B += (size_t)bz * (size_t)strideB;
  const size_t cbase = (size_t)bz * (size_t)strideC;

  const int bm   = blockIdx.y * 128;
  const int bn   = blockIdx.x * 128;
  const int tid  = threadIdx.x;
  const int wave = tid >> 5;
  const int lane = tid & 31;
  const int wm   = (wave >> 2) * 64;   // 2 waves along M
  const int wn   = (wave & 3) * 32;    // 4 waves along N
  const int lrow = lane & 15;
  const int lhalf = lane >> 4;

  // Per-thread tile-staging assignment: 2 x 16B chunks of A and of B.
  const int c0 = tid,        r0 = c0 >> 2, col0 = (c0 & 3) << 3;
  const int c1 = tid + 256,  r1 = c1 >> 2, col1 = (c1 & 3) << 3;
  const bf16* gA0 = &A[(size_t)(bm + r0) * lda + col0];
  const bf16* gA1 = &A[(size_t)(bm + r1) * lda + col1];
  const bf16* gB0 = &B[(size_t)(bn + r0) * ldb + col0];
  const bf16* gB1 = &B[(size_t)(bn + r1) * ldb + col1];
  const unsigned int lA0[2] = { lds_off_of(&shA[0][r0 * 32 + col0]), lds_off_of(&shA[1][r0 * 32 + col0]) };
  const unsigned int lA1[2] = { lds_off_of(&shA[0][r1 * 32 + col1]), lds_off_of(&shA[1][r1 * 32 + col1]) };
  const unsigned int lB0[2] = { lds_off_of(&shB[0][r0 * 32 + col0]), lds_off_of(&shB[1][r0 * 32 + col0]) };
  const unsigned int lB1[2] = { lds_off_of(&shB[0][r1 * 32 + col1]), lds_off_of(&shB[1][r1 * 32 + col1]) };

  auto issue_tile = [&](int k0, int buf) {
    async_load_b128(lA0[buf], gA0 + k0);
    async_load_b128(lA1[buf], gA1 + k0);
    async_load_b128(lB0[buf], gB0 + k0);
    async_load_b128(lB1[buf], gB1 + k0);
  };

  v8f acc[4][2];
#pragma unroll
  for (int i = 0; i < 4; ++i)
#pragma unroll
    for (int j = 0; j < 2; ++j) acc[i][j] = zero8();

  issue_tile(0, 0);
  int buf = 0;
  for (int k0 = 0; k0 < K; k0 += 32, buf ^= 1) {
    wait_async0();        // our slice of tile k0 is in LDS
    __syncthreads();      // whole tile present; previous buffer fully consumed
    if (k0 + 32 < K) issue_tile(k0 + 32, buf ^ 1);   // prefetch into other buffer

    // A fragment (16x32): lanes 0-15 hold K 0..15,
    // lanes 16-31 hold K 16..31 for row M = lane&15 (contiguous).
    v16bf af[4];
#pragma unroll
    for (int mi = 0; mi < 4; ++mi) {
      const bf16* p = &shA[buf][(wm + mi * 16 + lrow) * 32];
      af[mi] = make_v16(*(const v8bf*)(p + lhalf * 16),
                        *(const v8bf*)(p + lhalf * 16 + 8));
    }
    // B fragment (32x16): lane holds column n = lane&15 of B_wmma, i.e.
    // row (bn+n) of B-input, K range (lane>>4)*16 .. +15 (contiguous).
    v16bf bfr[2];
#pragma unroll
    for (int ni = 0; ni < 2; ++ni) {
      const bf16* p = &shB[buf][(wn + ni * 16 + lrow) * 32 + lhalf * 16];
      bfr[ni] = make_v16(*(const v8bf*)p, *(const v8bf*)(p + 8));
    }
#pragma unroll
    for (int mi = 0; mi < 4; ++mi)
#pragma unroll
      for (int ni = 0; ni < 2; ++ni)
        acc[mi][ni] = __builtin_amdgcn_wmma_f32_16x16x32_bf16(
            false, af[mi], false, bfr[ni], (short)0, acc[mi][ni], false, false);
  }

  // C fragment layout: lane L holds N = L&15, M = (L>>4)*8 + v.
#pragma unroll
  for (int mi = 0; mi < 4; ++mi)
#pragma unroll
    for (int ni = 0; ni < 2; ++ni) {
      const int rr = bm + wm + mi * 16 + lhalf * 8;
      const int cc = bn + wn + ni * 16 + lrow;
      if constexpr (F32OUT) {
        float* C = (float*)Cptr;
#pragma unroll
        for (int v = 0; v < 8; ++v)
          C[cbase + (size_t)(rr + v) * ldc + cc] = acc[mi][ni][v];
      } else {
        bf16* C = (bf16*)Cptr;
#pragma unroll
        for (int v = 0; v < 8; ++v)
          C[cbase + (size_t)(rr + v) * ldc + cc] = (bf16)acc[mi][ni][v];
      }
    }
}

// ---------------------------------------------------------------------------
// RoPE: read fp32 Q_lin/K_full in (B,S,NH*HD), rotate, write bf16 (B,NH,S,HD).
// One thread per (b,s,h, d<64) pair.
// ---------------------------------------------------------------------------
__global__ void rope_kernel(const float* __restrict__ Qlin,
                            const float* __restrict__ Kfull,
                            const int* __restrict__ pos_ids,
                            bf16* __restrict__ Qr, bf16* __restrict__ Kr)
{
  long idx = (long)blockIdx.x * blockDim.x + threadIdx.x;
  const long total = (long)BB * SS * NH * (HD / 2);
  if (idx >= total) return;
  const int i = (int)(idx & 63);          // rotary pair index
  long t = idx >> 6;
  const int h = (int)(t & (NH - 1)); t >>= 5;
  const int s = (int)(t & (SS - 1));
  const int b = (int)(t >> 11);

  const float pos = (float)pos_ids[b * SS + s];
  // inv_freq = 10000^(-2i/128)
  const float inv = __expf(-9.210340371976184f * (2.f * (float)i) / 128.f);
  float sn, cs;
  sincosf(pos * inv, &sn, &cs);

  const size_t src = (size_t)(b * SS + s) * HID + (size_t)h * HD;
  const size_t dst = ((size_t)(b * NH + h) * SS + s) * HD;

  float q1 = Qlin[src + i], q2 = Qlin[src + 64 + i];
  Qr[dst + i]      = (bf16)(q1 * cs - q2 * sn);
  Qr[dst + 64 + i] = (bf16)(q2 * cs + q1 * sn);

  float k1 = Kfull[src + i], k2 = Kfull[src + 64 + i];
  Kr[dst + i]      = (bf16)(k1 * cs - k2 * sn);
  Kr[dst + 64 + i] = (bf16)(k2 * cs + k1 * sn);
}

// ---------------------------------------------------------------------------
// Transpose V latent: (B, S, 2048) -> (B, 2048, S)   (both bf16)
// ---------------------------------------------------------------------------
__global__ void transpose_v(const bf16* __restrict__ Vlat, bf16* __restrict__ Vt) {
  long idx = (long)blockIdx.x * blockDim.x + threadIdx.x;
  long stride = (long)gridDim.x * blockDim.x;
  const long total = (long)BB * SS * (GG * FGD);
  for (; idx < total; idx += stride) {
    int n = (int)(idx & 2047);
    int s = (int)((idx >> 11) & 2047);
    int b = (int)(idx >> 22);
    Vt[((size_t)(b * 2048 + n) * SS) + s] = Vlat[idx];
  }
}

// ---------------------------------------------------------------------------
// Flash attention, 1 wave per 16-query tile per (b,h).
//   Qr/Kr: bf16 (B,NH,S,128);  Vt: bf16 (B, 2048v, 2048s)
//   Fused: bf16 (B, S, 8192) with the exact PALU reshape scramble.
// ---------------------------------------------------------------------------
__global__ __launch_bounds__(32) void flash_attn(
    const bf16* __restrict__ Qr, const bf16* __restrict__ Kr,
    const bf16* __restrict__ Vt, bf16* __restrict__ Fused)
{
  __shared__ __align__(16) bf16 sP[16 * 32];

  const int qt = blockIdx.x;          // query tile (16 queries)
  const int h  = blockIdx.y;
  const int b  = blockIdx.z;
  const int lane  = threadIdx.x;
  const int lrow  = lane & 15;
  const int lhalf = lane >> 4;
  const int q0 = qt * 16;
  const int g  = h >> 2;              // palu group
  const float scale = 0.08838834764831845f;  // 1/sqrt(128)

  // Q A-fragments (16 x 128 = 4 chunks of 16x32)
  const bf16* qbase = Qr + ((size_t)(b * NH + h) * SS + q0) * HD;
  v16bf qa[4];
#pragma unroll
  for (int c = 0; c < 4; ++c) {
    const bf16* p = qbase + (size_t)lrow * HD + c * 32;
    qa[c] = make_v16(*(const v8bf*)(p + lhalf * 16),
                     *(const v8bf*)(p + lhalf * 16 + 8));
  }

  float m_run[8], l_run[8];
#pragma unroll
  for (int v = 0; v < 8; ++v) { m_run[v] = -3.0e38f; l_run[v] = 0.f; }
  v8f o[16];
#pragma unroll
  for (int f = 0; f < 16; ++f) o[f] = zero8();

  const bf16* kbase = Kr + (size_t)(b * NH + h) * SS * HD;
  const bf16* vbase = Vt + ((size_t)b * 2048 + g * FGD) * SS;

  const int kend = q0 + 16;           // causal: keys 0 .. q0+15
  for (int kt = 0; kt < kend; kt += 32) {
    // ---- scores: two 16x16 fragments (keys kt..kt+15 and kt+16..kt+31) ----
    v8f s0 = zero8(), s1 = zero8();
#pragma unroll
    for (int c = 0; c < 4; ++c) {
      const bf16* p0 = kbase + (size_t)(kt + lrow) * HD + c * 32 + lhalf * 16;
      const bf16* p1 = kbase + (size_t)(kt + 16 + lrow) * HD + c * 32 + lhalf * 16;
      v16bf kb0 = make_v16(*(const v8bf*)p0, *(const v8bf*)(p0 + 8));
      v16bf kb1 = make_v16(*(const v8bf*)p1, *(const v8bf*)(p1 + 8));
      s0 = __builtin_amdgcn_wmma_f32_16x16x32_bf16(false, qa[c], false, kb0, (short)0, s0, false, false);
      s1 = __builtin_amdgcn_wmma_f32_16x16x32_bf16(false, qa[c], false, kb1, (short)0, s1, false, false);
    }

    // ---- scale, causal mask, online softmax (rows live in 16-lane halves) --
#pragma unroll
    for (int v = 0; v < 8; ++v) {
      const int qq  = q0 + lhalf * 8 + v;
      const int k0g = kt + lrow;
      const int k1g = kt + 16 + lrow;
      float a0 = (k0g > qq) ? -1.0e9f : s0[v] * scale;
      float a1 = (k1g > qq) ? -1.0e9f : s1[v] * scale;
      float mx = fmaxf(a0, a1);
#pragma unroll
      for (int off = 1; off < 16; off <<= 1) mx = fmaxf(mx, __shfl_xor(mx, off, 32));
      const float mnew = fmaxf(m_run[v], mx);
      const float corr = __expf(m_run[v] - mnew);
      const float p0 = __expf(a0 - mnew);
      const float p1 = __expf(a1 - mnew);
      s0[v] = p0; s1[v] = p1;
      float rs = p0 + p1;
#pragma unroll
      for (int off = 1; off < 16; off <<= 1) rs += __shfl_xor(rs, off, 32);
      l_run[v] = l_run[v] * corr + rs;
      m_run[v] = mnew;
#pragma unroll
      for (int f = 0; f < 16; ++f) o[f][v] *= corr;
    }

    // ---- C-layout -> A-layout for P via LDS (16x32 bf16) -------------------
#pragma unroll
    for (int v = 0; v < 8; ++v) {
      const int m = lhalf * 8 + v;
      sP[m * 32 + lrow]      = (bf16)s0[v];
      sP[m * 32 + 16 + lrow] = (bf16)s1[v];
    }
    __syncthreads();
    v16bf pa = make_v16(*(const v8bf*)&sP[lrow * 32 + lhalf * 16],
                        *(const v8bf*)&sP[lrow * 32 + lhalf * 16 + 8]);
    __syncthreads();

    // ---- O(16x256) += P(16x32) x V(32x256): 16 WMMAs ----------------------
#pragma unroll
    for (int f = 0; f < 16; ++f) {
      const bf16* p = vbase + (size_t)(f * 16 + lrow) * SS + kt + lhalf * 16;
      v16bf vb = make_v16(*(const v8bf*)p, *(const v8bf*)(p + 8));
      o[f] = __builtin_amdgcn_wmma_f32_16x16x32_bf16(false, pa, false, vb, (short)0, o[f], false, false);
    }
  }

  // ---- normalize and write fused buffer with PALU reshape scramble --------
#pragma unroll
  for (int v = 0; v < 8; ++v) {
    const float invl = 1.f / l_run[v];
    const int m  = lhalf * 8 + v;
    const int qq = q0 + m;
    const int hg = h & 3;
    const int idx = hg * SS + qq;          // hg*q + s
    const int t = idx >> 2, j = idx & 3;
    const size_t base = ((size_t)(b * SS) + t) * FH + (size_t)j * 2048 + (size_t)g * FGD;
#pragma unroll
    for (int f = 0; f < 16; ++f)
      Fused[base + f * 16 + lrow] = (bf16)(o[f][v] * invl);
  }
}

// ---------------------------------------------------------------------------
// Host-side orchestration
// ---------------------------------------------------------------------------
extern "C" void kernel_launch(void* const* d_in, const int* in_sizes, int n_in,
                              void* d_out, int out_size, void* d_ws, size_t ws_size,
                              hipStream_t stream) {
  (void)in_sizes; (void)n_in; (void)out_size; (void)ws_size;

  const float* hs   = (const float*)d_in[0];
  /* d_in[1] = attention_mask: causal -1e9 mask, applied analytically */
  const int*   pos  = (const int*)d_in[2];
  const float* Wq   = (const float*)d_in[3];
  const float* WVT  = (const float*)d_in[4];
  const float* U    = (const float*)d_in[5];
  const float* Wv   = (const float*)d_in[6];
  const float* Wo   = (const float*)d_in[7];
  float* out = (float*)d_out;

  // ---- workspace layout (bytes) ----
  char* ws = (char*)d_ws;
  size_t off = 0;
  auto alloc = [&](size_t bytes) { char* p = ws + off; off += (bytes + 255) & ~(size_t)255; return p; };

  bf16*  Wq_bf   = (bf16*)alloc((size_t)HID * HID * 2);        // 32 MB
  bf16*  WVT_bf  = (bf16*)alloc((size_t)GG * RK * HID * 2);    // 16 MB
  bf16*  Wv_bf   = (bf16*)alloc((size_t)GG * FGD * HID * 2);   // 16 MB
  bf16*  U_bf    = (bf16*)alloc((size_t)512 * 2048 * 2);       //  2 MB
  bf16*  Wo_bf   = (bf16*)alloc((size_t)HID * FH * 2);         // 64 MB
  bf16*  hs_bf   = (bf16*)alloc((size_t)BB * SS * HID * 2);    // 32 MB
  char*  qlin_fused = alloc((size_t)BB * SS * HID * 4);        // 64 MB (f32 Qlin, later bf16 Fused)
  float* Qlin  = (float*)qlin_fused;
  bf16*  Fused = (bf16*)qlin_fused;                            // alias: Qlin dead after RoPE
  float* Kfull = (float*)alloc((size_t)BB * SS * HID * 4);     // 64 MB
  bf16*  Klat  = (bf16*)alloc((size_t)BB * SS * GG * RK * 2);  // 16 MB
  bf16*  Vlat  = (bf16*)alloc((size_t)BB * SS * GG * FGD * 2); // 16 MB
  bf16*  Vt    = (bf16*)alloc((size_t)BB * 2048 * SS * 2);     // 16 MB
  bf16*  Qr    = (bf16*)alloc((size_t)BB * NH * SS * HD * 2);  // 32 MB
  bf16*  Kr    = (bf16*)alloc((size_t)BB * NH * SS * HD * 2);  // 32 MB

  const int CT = 256;
  auto cvt = [&](const float* s, bf16* d, long n) {
    int blocks = (int)((n + CT * 8 - 1) / (CT * 8));
    if (blocks > 16384) blocks = 16384;
    cvt_f32_bf16<<<blocks, CT, 0, stream>>>(s, d, n);
  };

  // 1) precision conversion
  cvt(hs,  hs_bf,  (long)BB * SS * HID);
  cvt(Wq,  Wq_bf,  (long)HID * HID);
  cvt(WVT, WVT_bf, (long)GG * RK * HID);
  cvt(Wv,  Wv_bf,  (long)GG * FGD * HID);
  cvt(U,   U_bf,   (long)512 * 2048);
  cvt(Wo,  Wo_bf,  (long)HID * FH);

  const int M = BB * SS;  // 4096

  // 2) Q_lin (f32) = hs @ Wq^T         [4096 x 4096 x 4096]
  gemm_bf16_nt<true><<<dim3(HID / 128, M / 128, 1), 256, 0, stream>>>(
      hs_bf, HID, 0, Wq_bf, HID, 0, (void*)Qlin, HID, 0, HID);

  // 3) K_lat (bf16) = hs @ WVT^T       [4096 x 2048 x 4096]
  gemm_bf16_nt<false><<<dim3(2048 / 128, M / 128, 1), 256, 0, stream>>>(
      hs_bf, HID, 0, WVT_bf, HID, 0, (void*)Klat, 2048, 0, HID);

  // 4) V_lat (bf16) = hs @ Wv^T        [4096 x 2048 x 4096]
  gemm_bf16_nt<false><<<dim3(2048 / 128, M / 128, 1), 256, 0, stream>>>(
      hs_bf, HID, 0, Wv_bf, HID, 0, (void*)Vlat, 2048, 0, HID);

  // 5) K reconstruction, batched over groups:
  //    Kfull[:, g*512 + d] = Klat[:, g*256:] @ (U[:, g*256:])^T   [4096 x 512 x 256]
  gemm_bf16_nt<true><<<dim3(512 / 128, M / 128, GG), 256, 0, stream>>>(
      Klat, 2048, RK, U_bf, 2048, RK, (void*)Kfull, HID, 512, RK);

  // 6) RoPE -> bf16 (B,NH,S,HD)
  {
    long total = (long)BB * SS * NH * (HD / 2);
    rope_kernel<<<(int)((total + CT - 1) / CT), CT, 0, stream>>>(Qlin, Kfull, pos, Qr, Kr);
  }

  // 7) V transpose -> (B, 2048v, 2048s)
  {
    long total = (long)BB * SS * 2048;
    int blocks = (int)((total + CT * 8 - 1) / (CT * 8));
    if (blocks > 16384) blocks = 16384;
    transpose_v<<<blocks, CT, 0, stream>>>(Vlat, Vt);
  }

  // 8) flash attention -> fused buffer (aliases dead Qlin)
  flash_attn<<<dim3(SS / 16, NH, BB), 32, 0, stream>>>(Qr, Kr, Vt, Fused);

  // 9) out (f32) = Fused @ Wo^T        [4096 x 4096 x 8192]
  gemm_bf16_nt<true><<<dim3(HID / 128, M / 128, 1), 256, 0, stream>>>(
      Fused, FH, 0, Wo_bf, FH, 0, (void*)out, HID, 0, FH);
}
